// RecurrentSpikingLayer_72292889526612
// MI455X (gfx1250) — compile-verified, matrix operand order checked by reading, not run on hardware
//
#include <hip/hip_runtime.h>

typedef __attribute__((ext_vector_type(16))) _Float16 v16h;
typedef __attribute__((ext_vector_type(8)))  _Float16 v8h;
typedef __attribute__((ext_vector_type(4)))  _Float16 v4h;
typedef __attribute__((ext_vector_type(8)))  float    v8f;
typedef __attribute__((ext_vector_type(4)))  float    v4f;

#define B_   128
#define T_   500
#define C_   700
#define CP_  704     // C padded to 32
#define N_   512
#define SEG  ((size_t)B_ * T_ * N_)     // 32,768,000 elements per output tensor

#define KC_W 22      // ceil(700/32) K-chunks for input GEMM
#define KC_V 16      // 512/32 K-chunks for recurrent GEMM

#define W_PACK_BYTES ((size_t)32 * KC_W * 32 * 16 * 2)   // 720,896
#define V_PACK_BYTES ((size_t)32 * KC_V * 32 * 16 * 2)   // 524,288

// ---------------------------------------------------------------------------
// Kernel 0: repack w (700x512) and v (512x512) f32 -> f16 in the exact
// per-lane WMMA B-fragment order for v_wmma_f32_16x16x32_f16.
// Fragment element e of lane L (n-tile nt, k-chunk kc):
//   n = nt*16 + (L & 15),  k = kc*32 + ((L>>4)*8) + (e<8 ? e : e+8)
// ---------------------------------------------------------------------------
__global__ void pack_weights_kernel(const float* __restrict__ w,
                                    const float* __restrict__ v,
                                    _Float16* __restrict__ w_pack,
                                    _Float16* __restrict__ v_pack) {
    const int tid = blockIdx.x * blockDim.x + threadIdx.x;
    const int WENT = 32 * KC_W * 32;   // 22528 fragment slots
    const int VENT = 32 * KC_V * 32;   // 16384 fragment slots
    if (tid < WENT) {
        const int lane = tid & 31;
        const int kc   = (tid >> 5) % KC_W;
        const int nt   = (tid >> 5) / KC_W;
        const int n    = nt * 16 + (lane & 15);
        const int kb   = kc * 32 + ((lane >> 4) << 3);
        _Float16* dst = w_pack + (size_t)tid * 16;
#pragma unroll
        for (int e = 0; e < 16; ++e) {
            const int k = kb + (e < 8 ? e : e + 8);
            dst[e] = (_Float16)((k < C_) ? w[(size_t)k * N_ + n] : 0.f);
        }
    } else if (tid < WENT + VENT) {
        const int t    = tid - WENT;
        const int lane = t & 31;
        const int kc   = (t >> 5) % KC_V;
        const int nt   = (t >> 5) / KC_V;
        const int n    = nt * 16 + (lane & 15);
        const int kb   = kc * 32 + ((lane >> 4) << 3);
        _Float16* dst = v_pack + (size_t)t * 16;
#pragma unroll
        for (int e = 0; e < 16; ++e) {
            const int k = kb + (e < 8 ? e : e + 8);
            dst[e] = (_Float16)v[(size_t)k * N_ + n];
        }
    }
}

// ---------------------------------------------------------------------------
// Kernel 1: h = inputs @ w   ([64000 x 700] x [700 x 512] -> [64000 x 512])
// One block (8 waves) per 16-row M-tile. The A-tile is loaded ONCE (NT loads,
// streamed: read exactly once), converted to f16 into a 22.5 KB LDS tile, and
// every wave computes 4 N-tiles from the shared A-fragment (4 independent
// WMMA accumulator chains per k-chunk). Result -> spk output segment.
// ---------------------------------------------------------------------------
__global__ __launch_bounds__(256)
void gemm_h_kernel(const float* __restrict__ x,        // [64000, 700]
                   const _Float16* __restrict__ w_pack,
                   float* __restrict__ h) {            // [64000, 512]
    __shared__ __align__(16) _Float16 xs[16][CP_];     // 22,528 B

    const int tid  = threadIdx.x;
    const int wave = tid >> 5;
    const int lane = tid & 31;
    const int m0   = blockIdx.x * 16;

    // --- cooperative A-tile load: 16 rows x 704 cols, f32 -> f16 ---
    const int NV4 = CP_ / 4;                   // 176 float4-groups per row
    for (int i = tid; i < 16 * NV4; i += 256) {
        const int row  = i / NV4;
        const int col4 = i - row * NV4;
        const int c    = col4 * 4;
        v4h hv;
        if (col4 < NV4 - 1) {                  // c+3 <= 699: in bounds
            const v4f f = __builtin_nontemporal_load(
                (const v4f*)(x + (size_t)(m0 + row) * C_ + c));
#pragma unroll
            for (int e = 0; e < 4; ++e) hv[e] = (_Float16)f[e];
        } else {                               // cols 700..703: zero pad
#pragma unroll
            for (int e = 0; e < 4; ++e) hv[e] = (_Float16)0.f;
        }
        *(v4h*)&xs[row][c] = hv;
    }
    __syncthreads();

    // --- WMMA: each wave owns n-tiles wave*4 .. wave*4+3 ---
    const int m  = lane & 15;
    const int kb = (lane >> 4) << 3;           // 0 or 8

    v8f acc[4] = {};
    for (int kc = 0; kc < KC_W; ++kc) {
        const v8h lo = *(const v8h*)&xs[m][kc * 32 + kb];
        const v8h hi = *(const v8h*)&xs[m][kc * 32 + kb + 16];
        v16h afrag;
#pragma unroll
        for (int e = 0; e < 8; ++e) { afrag[e] = lo[e]; afrag[8 + e] = hi[e]; }
#pragma unroll
        for (int j = 0; j < 4; ++j) {
            const int nt = wave * 4 + j;
            const v16h bfrag = *(const v16h*)(w_pack +
                                ((size_t)(nt * KC_W + kc) * 32 + lane) * 16);
            acc[j] = __builtin_amdgcn_wmma_f32_16x16x32_f16(
                         false, afrag, false, bfrag, (short)0, acc[j], false, false);
        }
    }

    const int row0 = m0 + ((lane >> 4) << 3);
#pragma unroll
    for (int j = 0; j < 4; ++j) {
        const int col = (wave * 4 + j) * 16 + (lane & 15);
#pragma unroll
        for (int r = 0; r < 8; ++r)
            h[(size_t)(row0 + r) * N_ + col] = acc[j][r];   // RT: re-read by scan
    }
}

// ---------------------------------------------------------------------------
// Kernel 2: the 499-step recurrent scan.
// 8 blocks x 1024 threads (32 waves). Block = 16 batch rows x all 512 neurons.
// Wave w owns N-tile w (16 columns). Spikes staged in LDS as f16; recurrent
// out@v runs as 16 chained WMMAs per wave per step, accumulating onto
// C = alpha*syn + h_t.  h lives in the spk output segment; h[t+1] is loaded
// into registers BEFORE spk[t+1] overwrites it (same thread, same addresses).
// All four output streams use non-temporal stores: written once, never
// re-read, keeping L2 reserved for h and v_pack.
// ---------------------------------------------------------------------------
__global__ __launch_bounds__(1024)
void scan_kernel(const _Float16* __restrict__ v_pack,
                 const float* __restrict__ a_p,  const float* __restrict__ b_p,
                 const float* __restrict__ er_p, const float* __restrict__ theta_p,
                 const float* __restrict__ g_p,  const float* __restrict__ eta_p,
                 const float* __restrict__ th_p, const float* __restrict__ alpha_p,
                 const float* __restrict__ wj_p,
                 float* out) {                        // NOT restrict: spk aliases h
    __shared__ __align__(16) _Float16 out_lds[16][N_];   // 16 KB spike tile

    const int wave  = threadIdx.x >> 5;     // 0..31 == n-tile
    const int lane  = threadIdx.x & 31;
    const int n     = wave * 16 + (lane & 15);
    const int rowhi = (lane >> 4) << 3;     // 0 or 8 (acc-layout row offset)
    const int b0    = blockIdx.x * 16;

    // per-lane neuron parameters (fixed n per lane)
    const float thv = th_p[n],   alphav = alpha_p[n], av = a_p[n], bv = b_p[n];
    const float erv = er_p[n],   thetav = theta_p[n], gv = g_p[n];
    const float etav = eta_p[n], wjv = wj_p[n];

    float* syn_o = out;
    float* mem_o = out + SEG;
    float* rec_o = out + 2 * SEG;
    float* spk_o = out + 3 * SEG;           // also where h was staged
    float* h     = spk_o;

    size_t base[8];
    float syn[8], mem[8], rec[8], hcur[8];
#pragma unroll
    for (int r = 0; r < 8; ++r) {
        base[r] = (size_t)(b0 + rowhi + r) * T_ * N_ + n;
        syn[r] = 0.f; mem[r] = 0.f; rec[r] = 0.f;
        hcur[r] = h[base[r]];               // h[b, t=0]
    }

    for (int t = 0; t < T_ - 1; ++t) {
        // --- spikes from current mem, staged to LDS as f16 A-matrix ---
        float outv[8];
#pragma unroll
        for (int r = 0; r < 8; ++r) {
            outv[r] = (mem[r] - thv) > 0.f ? 1.f : 0.f;
            out_lds[rowhi + r][n] = (_Float16)outv[r];
        }

        // --- acc = alpha*syn + h_t, then chain out@v WMMAs onto it ---
        v8f acc;
#pragma unroll
        for (int r = 0; r < 8; ++r)
            acc[r] = alphav * syn[r] + hcur[r];

        __syncthreads();

        const int m  = lane & 15;
        const int kb = (lane >> 4) << 3;
        for (int kc = 0; kc < KC_V; ++kc) {
            const v8h lo = *(const v8h*)&out_lds[m][kc * 32 + kb];
            const v8h hi = *(const v8h*)&out_lds[m][kc * 32 + kb + 16];
            v16h afrag;
#pragma unroll
            for (int e = 0; e < 8; ++e) { afrag[e] = lo[e]; afrag[8 + e] = hi[e]; }
            const v16h bfrag = *(const v16h*)(v_pack +
                                ((size_t)(wave * KC_V + kc) * 32 + lane) * 16);
            acc = __builtin_amdgcn_wmma_f32_16x16x32_f16(
                      false, afrag, false, bfrag, (short)0, acc, false, false);
        }
        __syncthreads();

        // --- prefetch h[t+1] BEFORE spk[t+1] store clobbers it ---
        float hnext[8];
#pragma unroll
        for (int r = 0; r < 8; ++r)
            hnext[r] = h[base[r] + (size_t)(t + 1) * N_];

        // --- neuron dynamics + output stores at t+1 (non-temporal) ---
#pragma unroll
        for (int r = 0; r < 8; ++r) {
            const float memv = mem[r], recv = rec[r], synv = syn[r];
            float nm = memv + 0.05f * (memv * (memv - 0.5f) + etav - recv
                                       + gv * synv * (erv - memv));
            const float drec = 1e-3f * (av * (bv * memv - recv));
            const bool  sp   = outv[r] > 0.f;
            const float nr   = sp ? (recv + wjv) : (recv + drec);
            nm = sp ? (thetav * (memv - thv)) : nm;

            syn[r] = acc[r]; mem[r] = nm; rec[r] = nr;

            const size_t idx = base[r] + (size_t)(t + 1) * N_;
            __builtin_nontemporal_store(acc[r],  &syn_o[idx]);
            __builtin_nontemporal_store(nm,      &mem_o[idx]);
            __builtin_nontemporal_store(nr,      &rec_o[idx]);
            __builtin_nontemporal_store(outv[r], &spk_o[idx]);
            if (t == 0) {   // prepend zero state (h[b,0] already consumed)
                __builtin_nontemporal_store(0.f, &syn_o[base[r]]);
                __builtin_nontemporal_store(0.f, &mem_o[base[r]]);
                __builtin_nontemporal_store(0.f, &rec_o[base[r]]);
                __builtin_nontemporal_store(0.f, &spk_o[base[r]]);
            }
        }
#pragma unroll
        for (int r = 0; r < 8; ++r) hcur[r] = hnext[r];
    }
}

// ---------------------------------------------------------------------------
extern "C" void kernel_launch(void* const* d_in, const int* in_sizes, int n_in,
                              void* d_out, int out_size, void* d_ws, size_t ws_size,
                              hipStream_t stream) {
    const float* x      = (const float*)d_in[0];   // inputs [128,500,700]
    const float* w      = (const float*)d_in[1];   // [700,512]
    const float* v      = (const float*)d_in[2];   // [512,512]
    const float* a_p    = (const float*)d_in[3];
    const float* b_p    = (const float*)d_in[4];
    const float* er_p   = (const float*)d_in[5];
    const float* th_a_p = (const float*)d_in[6];   // theta
    const float* g_p    = (const float*)d_in[7];
    const float* eta_p  = (const float*)d_in[8];
    const float* th_p   = (const float*)d_in[9];   // th
    const float* alp_p  = (const float*)d_in[10];  // alpha
    const float* wj_p   = (const float*)d_in[11];  // w_jump

    float* out = (float*)d_out;
    _Float16* w_pack = (_Float16*)d_ws;                              // 720,896 B
    _Float16* v_pack = (_Float16*)((char*)d_ws + W_PACK_BYTES);      // 524,288 B

    // Phase 0: repack weights into WMMA fragment order (f16)
    const int pack_threads = 32 * KC_W * 32 + 32 * KC_V * 32;        // 38,912
    pack_weights_kernel<<<(pack_threads + 255) / 256, 256, 0, stream>>>(
        w, v, w_pack, v_pack);

    // Phase 1: h = inputs @ w, staged into the spk output segment
    gemm_h_kernel<<<B_ * T_ / 16, 256, 0, stream>>>(
        x, w_pack, out + 3 * SEG);

    // Phase 2: recurrent scan (8 blocks x 32 waves)
    scan_kernel<<<B_ / 16, 1024, 0, stream>>>(
        v_pack, a_p, b_p, er_p, th_a_p, g_p, eta_p, th_p, alp_p, wj_p, out);
}
